// NanoJEPA_86638080295005
// MI455X (gfx1250) — matrix operand order repaired and checked by reading, not compiled, not run on hardware
//
#include <hip/hip_runtime.h>

typedef __attribute__((ext_vector_type(16))) __bf16 v16bf;
typedef __attribute__((ext_vector_type(8)))  __bf16 v8bf;
typedef __attribute__((ext_vector_type(8)))  float  v8f;
typedef __attribute__((ext_vector_type(4)))  unsigned int u32x4;
typedef __attribute__((ext_vector_type(4)))  int i32x4;
typedef __attribute__((ext_vector_type(8)))  int i32x8;

constexpr int LNUM  = 8;
constexpr int BNUM  = 2;
constexpr int TSEQ  = 1024;
constexpr int CDIM  = 1024;
constexpr int HNUM  = 16;
constexpr int HD    = 64;           // CDIM / HNUM
constexpr int VOCAB = 50257;
constexpr int ROWS  = BNUM * TSEQ;  // 2048 tokens

union Frag16 { v8bf h[2]; v16bf v; };
union FragI  { int  i[8]; v16bf v; };

__device__ __forceinline__ v8f wmma_bf16(v16bf a, v16bf b, v8f c) {
  return __builtin_amdgcn_wmma_f32_16x16x32_bf16(false, a, false, b, (short)0, c,
                                                 false, false);
}

#if defined(__has_builtin)
#if __has_builtin(__builtin_amdgcn_tensor_load_to_lds)
#define HAVE_TDM 1
#endif
#endif

#ifdef HAVE_TDM
// Issue one TDM 2-D tile load: tile_dim0=32 (contiguous K), tile_dim1=128 rows,
// row stride = K elements, 2-byte elements, compacted row-major into LDS.
__device__ __forceinline__ void tdm_load_a_tile(const __bf16* gptr,
                                                unsigned lds_off, int K, int M) {
  unsigned long long ga = (unsigned long long)(uintptr_t)gptr;
  u32x4 g0;
  g0[0] = 1u;                                      // count=1, user-mode, no gather
  g0[1] = lds_off;                                 // lds_addr (bytes)
  g0[2] = (unsigned)(ga & 0xffffffffu);            // global_addr[31:0]
  g0[3] = (unsigned)((ga >> 32) & 0x01ffffffu)     // global_addr[56:32]
          | (2u << 30);                            // type = 2 ("image")
  i32x8 g1;
  g1[0] = (1 << 16);                               // wg_mask=0, data_size=1 (2 B)
  g1[1] = (int)((unsigned)(K & 0xffff) << 16);     // tensor_dim0[15:0] @bit48
  g1[2] = (int)(((unsigned)K >> 16) & 0xffffu)     // tensor_dim0[31:16]
          | (int)((unsigned)(M & 0xffff) << 16);   // tensor_dim1[15:0]
  g1[3] = (int)(((unsigned)M >> 16) & 0xffffu)     // tensor_dim1[31:16]
          | (32 << 16);                            // tile_dim0 = 32
  g1[4] = 128;                                     // tile_dim1 = 128, tile_dim2 = 0
  g1[5] = K;                                       // tensor_dim0_stride[31:0]
  g1[6] = 0;                                       // stride hi + dim1_stride lo
  g1[7] = 0;
  i32x4 g2 = {};
  i32x4 g3 = {};
#if __clang_major__ >= 23
  i32x8 g4 = {};
  __builtin_amdgcn_tensor_load_to_lds(g0, g1, g2, g3, g4, 0);
#else
  __builtin_amdgcn_tensor_load_to_lds(g0, g1, g2, g3, 0);
#endif
}
#endif

// ---------------------------------------------------------------- cvt f32->bf16
__global__ void cvt_kernel(const float* __restrict__ in, __bf16* __restrict__ out,
                           size_t n) {
  size_t i = (size_t)blockIdx.x * blockDim.x + threadIdx.x;
  size_t stride = (size_t)gridDim.x * blockDim.x;
  for (; i < n; i += stride) out[i] = (__bf16)in[i];
}

// ---------------------------------------------------------------- embeddings
__global__ void embed_kernel(const int* __restrict__ ids,
                             const float* __restrict__ wte,
                             const float* __restrict__ wpe,
                             float* __restrict__ x) {
  int rt = blockIdx.x;           // 0..ROWS-1
  int t  = rt % TSEQ;
  int id = ids[rt];
  for (int c = threadIdx.x; c < CDIM; c += blockDim.x)
    x[(size_t)rt * CDIM + c] = wte[(size_t)id * CDIM + c] + wpe[(size_t)t * CDIM + c];
}

// ---------------------------------------------------------------- layernorm -> bf16
__global__ __launch_bounds__(256) void ln_kernel(const float* __restrict__ x,
                                                 const float* __restrict__ w,
                                                 const float* __restrict__ b,
                                                 __bf16* __restrict__ out) {
  __shared__ float s1[256], s2[256];
  int row = blockIdx.x;
  const float* xr = x + (size_t)row * CDIM;
  float a = 0.f, q = 0.f;
  for (int c = threadIdx.x; c < CDIM; c += 256) {
    float v = xr[c];
    a += v; q += v * v;
  }
  s1[threadIdx.x] = a; s2[threadIdx.x] = q;
  __syncthreads();
  for (int s = 128; s > 0; s >>= 1) {
    if (threadIdx.x < s) {
      s1[threadIdx.x] += s1[threadIdx.x + s];
      s2[threadIdx.x] += s2[threadIdx.x + s];
    }
    __syncthreads();
  }
  float mean = s1[0] * (1.f / CDIM);
  float var  = s2[0] * (1.f / CDIM) - mean * mean;
  float r    = rsqrtf(var + 1e-5f);
  for (int c = threadIdx.x; c < CDIM; c += 256)
    out[(size_t)row * CDIM + c] = (__bf16)((xr[c] - mean) * r * w[c] + b[c]);
}

// ---------------------------------------------------------------- WMMA GEMM
// C[M,N] = A[M,K](bf16) x B(bf16) (+bias, epilogue)
// BT=false: B is [K,N] row-major.  BT=true: B is [N,K] row-major (tied lm_head).
// MODE 0: +bias            -> bf16 out
// MODE 1: +bias, erf-GELU  -> bf16 out
// MODE 2: +bias, +residual -> f32 out
// MODE 3: (+bias if !null) -> f32 out
template <bool BT, int MODE>
__global__ __launch_bounds__(256) void gemm_kernel(
    const __bf16* __restrict__ A, const __bf16* __restrict__ Bm,
    const float* __restrict__ bias, const float* __restrict__ resid,
    float* __restrict__ outF, __bf16* __restrict__ outB,
    int M, int N, int K) {
  __shared__ __bf16 sA[128 * 32];   // [row][k]
  __shared__ int    sB[16 * 128];   // k-pair packed: dword (k/2,n) = {B[k][n],B[k+1][n]}
  __bf16* sBh = (__bf16*)sB;

  const int t    = threadIdx.x;
  const int lane = t & 31;
  const int lo   = lane & 15;
  const int kh   = lane >> 4;
  const int wid  = t >> 5;
  const int waveM = wid >> 1;       // 0..3
  const int waveN = wid & 1;        // 0..1
  const int m0 = blockIdx.y * 128;
  const int n0 = blockIdx.x * 128;
#ifdef HAVE_TDM
  const unsigned sA_off = (unsigned)(uintptr_t)&sA[0];
#endif

  v8f acc[2][4];
#pragma unroll
  for (int mi = 0; mi < 2; ++mi)
#pragma unroll
    for (int nj = 0; nj < 4; ++nj) acc[mi][nj] = (v8f){};

  for (int k0 = 0; k0 < K; k0 += 32) {
#ifdef HAVE_TDM
    // A tile 128x32 via Tensor Data Mover (one descriptor, wave 0 issues).
    if (wid == 0)
      tdm_load_a_tile(A + (size_t)m0 * K + k0, sA_off, K, M);
#else
    { // A tile 128x32 : 16 bf16 per thread
      int row = t >> 1, half = t & 1;
      const v8bf* src = (const v8bf*)(A + (size_t)(m0 + row) * K + k0 + half * 16);
      v8bf* dst = (v8bf*)(sA + row * 32 + half * 16);
      dst[0] = src[0];
      dst[1] = src[1];
    }
#endif
    if (!BT) { // B tile 32x128 from [K,N]
      int kr = t >> 3, nb = (t & 7) * 16;
      const __bf16* src = Bm + (size_t)(k0 + kr) * N + n0 + nb;
      if (k0 + 32 < K)
        __builtin_prefetch(src + (size_t)32 * N, 0, 0);   // next K-step weight tile
      __bf16 tmp[16];
      if (n0 + nb + 16 <= N) {
        v8bf a0 = ((const v8bf*)src)[0];
        v8bf a1 = ((const v8bf*)src)[1];
#pragma unroll
        for (int j = 0; j < 8; ++j) { tmp[j] = a0[j]; tmp[8 + j] = a1[j]; }
      } else {
#pragma unroll
        for (int j = 0; j < 16; ++j)
          tmp[j] = (n0 + nb + j < N) ? src[j] : (__bf16)0.f;
      }
#pragma unroll
      for (int j = 0; j < 16; ++j)
        sBh[(kr >> 1) * 256 + (nb + j) * 2 + (kr & 1)] = tmp[j];
    } else { // B tile from [N,K] (transposed contraction)
      int n = t >> 1, half = t & 1;
      __bf16 tmp[16];
      if (n0 + n < N) {
        const __bf16* srcp = Bm + (size_t)(n0 + n) * K + k0 + half * 16;
        if (k0 + 32 < K) __builtin_prefetch(srcp + 32, 0, 0);
        const v8bf* src = (const v8bf*)srcp;
        v8bf a0 = src[0], a1 = src[1];
#pragma unroll
        for (int j = 0; j < 8; ++j) { tmp[j] = a0[j]; tmp[8 + j] = a1[j]; }
      } else {
#pragma unroll
        for (int j = 0; j < 16; ++j) tmp[j] = (__bf16)0.f;
      }
#pragma unroll
      for (int j = 0; j < 16; ++j) {
        int k = half * 16 + j;
        sBh[(k >> 1) * 256 + n * 2 + (k & 1)] = tmp[j];
      }
    }
#ifdef HAVE_TDM
    if (wid == 0) __builtin_amdgcn_s_wait_tensorcnt(0);
#endif
    __syncthreads();

    Frag16 af[2];
#pragma unroll
    for (int mi = 0; mi < 2; ++mi) {
      int row = waveM * 32 + mi * 16 + lo;
      af[mi].h[0] = *(const v8bf*)(sA + row * 32 + kh * 8);
      af[mi].h[1] = *(const v8bf*)(sA + row * 32 + 16 + kh * 8);
    }
    FragI bf[4];
#pragma unroll
    for (int nj = 0; nj < 4; ++nj) {
      int n = waveN * 64 + nj * 16 + lo;
#pragma unroll
      for (int j = 0; j < 8; ++j) bf[nj].i[j] = sB[((j >> 2) * 8 + kh * 4 + (j & 3)) * 128 + n];
    }
#pragma unroll
    for (int mi = 0; mi < 2; ++mi)
#pragma unroll
      for (int nj = 0; nj < 4; ++nj)
        acc[mi][nj] = wmma_bf16(af[mi].v, bf[nj].v, acc[mi][nj]);
    __syncthreads();
  }

  // epilogue
#pragma unroll
  for (int nj = 0; nj < 4; ++nj) {
    int col = n0 + waveN * 64 + nj * 16 + lo;
    if (col >= N) continue;
    float bv = (bias != nullptr) ? bias[col] : 0.f;
#pragma unroll
    for (int mi = 0; mi < 2; ++mi) {
#pragma unroll
      for (int i = 0; i < 8; ++i) {
        int row = m0 + waveM * 32 + mi * 16 + i + 8 * kh;
        size_t idx = (size_t)row * N + col;
        float v = acc[mi][nj][i] + bv;
        if (MODE == 0) {
          outB[idx] = (__bf16)v;
        } else if (MODE == 1) {
          float g = 0.5f * v * (1.f + erff(v * 0.70710678118654752f));
          outB[idx] = (__bf16)g;
        } else if (MODE == 2) {
          outF[idx] = v + resid[idx];
        } else {
          outF[idx] = v;
        }
      }
    }
  }
}

// ---------------------------------------------------------------- flash attention
// One wave per (b, h, 16-query tile). No mask (matches reference). qkv: [B,T,3C] bf16.
__global__ __launch_bounds__(32) void attn_kernel(const __bf16* __restrict__ qkv,
                                                  __bf16* __restrict__ y) {
  __shared__ __bf16 vT[HD * 32];   // [dim][key]  (4 KB)
  __shared__ __bf16 pT[16 * 32];   // [row][key]  (1 KB)

  const int bid = blockIdx.x;
  const int qb = bid & 63;
  const int h  = (bid >> 6) & (HNUM - 1);
  const int b  = bid >> 10;
  const int lane = threadIdx.x;
  const int lo = lane & 15, kh = lane >> 4;
  const size_t rs = 3 * CDIM;           // qkv row stride (elements)
  const size_t base = (size_t)b * TSEQ;
  const int t0 = qb * 16;
  const float scale = 0.125f;           // 1/sqrt(64)

  Frag16 aq0, aq1;
  {
    const __bf16* qp = qkv + (base + t0 + lo) * rs + h * HD;
    aq0.h[0] = *(const v8bf*)(qp + kh * 8);
    aq0.h[1] = *(const v8bf*)(qp + 16 + kh * 8);
    aq1.h[0] = *(const v8bf*)(qp + 32 + kh * 8);
    aq1.h[1] = *(const v8bf*)(qp + 48 + kh * 8);
  }

  float m[8], l[8];
  v8f o[4];
#pragma unroll
  for (int i = 0; i < 8; ++i) { m[i] = -1e30f; l[i] = 0.f; }
#pragma unroll
  for (int nt = 0; nt < 4; ++nt) o[nt] = (v8f){};

  for (int jb = 0; jb < TSEQ; jb += 32) {
    { // stage V^T: lane = key
      const __bf16* vp = qkv + (base + jb + lane) * rs + 2 * CDIM + h * HD;
      v8bf vr[8];
#pragma unroll
      for (int d8 = 0; d8 < 8; ++d8) vr[d8] = *(const v8bf*)(vp + d8 * 8);
#pragma unroll
      for (int d8 = 0; d8 < 8; ++d8)
#pragma unroll
        for (int j = 0; j < 8; ++j) vT[(d8 * 8 + j) * 32 + lane] = vr[d8][j];
    }

    // S = Q K^T for two 16-key sub-blocks
    v8f s0, s1;
#pragma unroll
    for (int sub = 0; sub < 2; ++sub) {
      const __bf16* kp = qkv + (base + jb + sub * 16 + lo) * rs + CDIM + h * HD;
      Frag16 kf0, kf1;
      kf0.h[0] = *(const v8bf*)(kp + kh * 8);
      kf0.h[1] = *(const v8bf*)(kp + 16 + kh * 8);
      kf1.h[0] = *(const v8bf*)(kp + 32 + kh * 8);
      kf1.h[1] = *(const v8bf*)(kp + 48 + kh * 8);
      v8f s = (v8f){};
      s = wmma_bf16(aq0.v, kf0.v, s);
      s = wmma_bf16(aq1.v, kf1.v, s);
      if (sub == 0) s0 = s; else s1 = s;
    }

    // online softmax; rows i + 8*kh live in this lane's half-wave
#pragma unroll
    for (int i = 0; i < 8; ++i) {
      float smax = fmaxf(s0[i] * scale, s1[i] * scale);
      smax = fmaxf(smax, __shfl_xor(smax, 8, 16));
      smax = fmaxf(smax, __shfl_xor(smax, 4, 16));
      smax = fmaxf(smax, __shfl_xor(smax, 2, 16));
      smax = fmaxf(smax, __shfl_xor(smax, 1, 16));
      float mn = fmaxf(m[i], smax);
      float ef = __expf(m[i] - mn);
      m[i] = mn;
      float p0 = __expf(s0[i] * scale - mn);
      float p1 = __expf(s1[i] * scale - mn);
      s0[i] = p0; s1[i] = p1;
      float r = p0 + p1;
      r += __shfl_xor(r, 8, 16);
      r += __shfl_xor(r, 4, 16);
      r += __shfl_xor(r, 2, 16);
      r += __shfl_xor(r, 1, 16);
      l[i] = l[i] * ef + r;
#pragma unroll
      for (int nt = 0; nt < 4; ++nt) o[nt][i] *= ef;
    }

    // stage P (D layout -> A layout via LDS)
#pragma unroll
    for (int i = 0; i < 8; ++i) {
      int row = i + 8 * kh;
      pT[row * 32 + lo]      = (__bf16)s0[i];
      pT[row * 32 + 16 + lo] = (__bf16)s1[i];
    }
    __syncthreads();

    Frag16 pf;
    pf.h[0] = *(const v8bf*)(pT + lo * 32 + kh * 8);
    pf.h[1] = *(const v8bf*)(pT + lo * 32 + 16 + kh * 8);
#pragma unroll
    for (int nt = 0; nt < 4; ++nt) {
      Frag16 vf;
      vf.h[0] = *(const v8bf*)(vT + (nt * 16 + lo) * 32 + kh * 8);
      vf.h[1] = *(const v8bf*)(vT + (nt * 16 + lo) * 32 + 16 + kh * 8);
      o[nt] = wmma_bf16(pf.v, vf.v, o[nt]);
    }
    __syncthreads();
  }

#pragma unroll
  for (int nt = 0; nt < 4; ++nt)
#pragma unroll
    for (int i = 0; i < 8; ++i) {
      int row = i + 8 * kh;
      float val = o[nt][i] / l[i];
      y[(base + t0 + row) * CDIM + h * HD + nt * 16 + lo] = (__bf16)val;
    }
}

// ---------------------------------------------------------------- launch
extern "C" void kernel_launch(void* const* d_in, const int* in_sizes, int n_in,
                              void* d_out, int out_size, void* d_ws, size_t ws_size,
                              hipStream_t stream) {
  (void)in_sizes; (void)n_in; (void)out_size; (void)ws_size;
  const int*   ids    = (const int*)d_in[0];
  const float* wte    = (const float*)d_in[1];
  const float* wpe    = (const float*)d_in[2];
  const float* ln1_w  = (const float*)d_in[3];
  const float* ln1_b  = (const float*)d_in[4];
  const float* attn_w = (const float*)d_in[5];
  const float* attn_b = (const float*)d_in[6];
  const float* proj_w = (const float*)d_in[7];
  const float* proj_b = (const float*)d_in[8];
  const float* ln2_w  = (const float*)d_in[9];
  const float* ln2_b  = (const float*)d_in[10];
  const float* fc_w   = (const float*)d_in[11];
  const float* fc_b   = (const float*)d_in[12];
  const float* fc2_w  = (const float*)d_in[13];
  const float* fc2_b  = (const float*)d_in[14];
  const float* lnf_w  = (const float*)d_in[15];
  const float* lnf_b  = (const float*)d_in[16];

  char* wsp = (char*)d_ws;
  size_t off = 0;
  auto alloc = [&](size_t bytes) -> char* {
    char* p = wsp + off;
    off = (off + bytes + 255) & ~(size_t)255;
    return p;
  };
  const size_t nWte  = (size_t)VOCAB * CDIM;
  const size_t nAttn = (size_t)LNUM * CDIM * 3 * CDIM;
  const size_t nProj = (size_t)LNUM * CDIM * CDIM;
  const size_t nFc   = (size_t)LNUM * CDIM * 4 * CDIM;
  const size_t nFc2  = (size_t)LNUM * 4 * CDIM * CDIM;

  __bf16* wteb   = (__bf16*)alloc(nWte * 2);
  __bf16* attnwb = (__bf16*)alloc(nAttn * 2);
  __bf16* projwb = (__bf16*)alloc(nProj * 2);
  __bf16* fcwb   = (__bf16*)alloc(nFc * 2);
  __bf16* fc2wb  = (__bf16*)alloc(nFc2 * 2);
  float*  x      = (float*)alloc((size_t)ROWS * CDIM * 4);
  __bf16* hb     = (__bf16*)alloc((size_t)ROWS * CDIM * 2);
  __bf16* qkvb   = (__bf16*)alloc((size_t)ROWS * 3 * CDIM * 2);
  __bf16* yb     = (__bf16*)alloc((size_t)ROWS * CDIM * 2);
  __bf16* ub     = (__bf16*)alloc((size_t)ROWS * 4 * CDIM * 2);

  dim3 blk256(256), blk32(32);
  cvt_kernel<<<dim3(4096), blk256, 0, stream>>>(wte, wteb, nWte);
  cvt_kernel<<<dim3(4096), blk256, 0, stream>>>(attn_w, attnwb, nAttn);
  cvt_kernel<<<dim3(4096), blk256, 0, stream>>>(proj_w, projwb, nProj);
  cvt_kernel<<<dim3(4096), blk256, 0, stream>>>(fc_w, fcwb, nFc);
  cvt_kernel<<<dim3(4096), blk256, 0, stream>>>(fc2_w, fc2wb, nFc2);

  embed_kernel<<<dim3(ROWS), blk256, 0, stream>>>(ids, wte, wpe, x);

  const dim3 gQkv(3 * CDIM / 128, ROWS / 128);
  const dim3 gProj(CDIM / 128, ROWS / 128);
  const dim3 gFc(4 * CDIM / 128, ROWS / 128);
  const dim3 gAttn(BNUM * HNUM * (TSEQ / 16));

  for (int l = 0; l < LNUM; ++l) {
    ln_kernel<<<dim3(ROWS), blk256, 0, stream>>>(x, ln1_w + (size_t)l * CDIM,
                                                 ln1_b + (size_t)l * CDIM, hb);
    gemm_kernel<false, 0><<<gQkv, blk256, 0, stream>>>(
        hb, attnwb + (size_t)l * CDIM * 3 * CDIM, attn_b + (size_t)l * 3 * CDIM,
        nullptr, nullptr, qkvb, ROWS, 3 * CDIM, CDIM);
    attn_kernel<<<gAttn, blk32, 0, stream>>>(qkvb, yb);
    gemm_kernel<false, 2><<<gProj, blk256, 0, stream>>>(
        yb, projwb + (size_t)l * CDIM * CDIM, proj_b + (size_t)l * CDIM,
        x, x, nullptr, ROWS, CDIM, CDIM);
    ln_kernel<<<dim3(ROWS), blk256, 0, stream>>>(x, ln2_w + (size_t)l * CDIM,
                                                 ln2_b + (size_t)l * CDIM, hb);
    gemm_kernel<false, 1><<<gFc, blk256, 0, stream>>>(
        hb, fcwb + (size_t)l * CDIM * 4 * CDIM, fc_b + (size_t)l * 4 * CDIM,
        nullptr, nullptr, ub, ROWS, 4 * CDIM, CDIM);
    gemm_kernel<false, 2><<<gProj, blk256, 0, stream>>>(
        ub, fc2wb + (size_t)l * 4 * CDIM * CDIM, fc2_b + (size_t)l * CDIM,
        x, x, nullptr, ROWS, CDIM, 4 * CDIM);
  }

  ln_kernel<<<dim3(ROWS), blk256, 0, stream>>>(x, lnf_w, lnf_b, hb);
  gemm_kernel<true, 3><<<dim3((VOCAB + 127) / 128, ROWS / 128), blk256, 0, stream>>>(
      hb, wteb, nullptr, nullptr, (float*)d_out, nullptr, ROWS, VOCAB, CDIM);
}
